// MCSPN_55825984913487
// MI455X (gfx1250) — matrix operand order repaired and checked by reading, not compiled, not run on hardware
//
#include <hip/hip_runtime.h>
#include <hip/hip_bf16.h>
#include <stddef.h>

// ---------------------------------------------------------------------------
// MCSPN on MI455X (gfx1250): conv3x3+BN+ReLU+conv1x1+softmax fused via bf16
// WMMA (v_wmma_f32_16x16x32_bf16), then 4 bandwidth-bound propagation steps.
// ---------------------------------------------------------------------------

typedef __attribute__((ext_vector_type(16))) __bf16 v16bf;
typedef __attribute__((ext_vector_type(8)))  __bf16 v8bf;
typedef __attribute__((ext_vector_type(8)))  float  v8f;

#define CIN   256
#define MID   128
#define KCLS  19
#define HH    128
#define WW    256
#define BB    8
#define KTOT  2304          // CIN * 9, ordered chunk-major: [c8][tap][c%32]
#define M2PAD 80            // 76 out channels padded to 80 (5 WMMA M-tiles)

__device__ __forceinline__ unsigned short f2bf(float f) {
    unsigned u = __float_as_uint(f);
    unsigned r = ((u >> 16) & 1u) + 0x7fffu;   // round-to-nearest-even
    return (unsigned short)((u + r) >> 16);
}

// --------------------- prep: fold BN into w1, reorder, bf16 -----------------
__global__ void mcspn_prep_w1(const float* __restrict__ w1,
                              const float* __restrict__ gamma,
                              const float* __restrict__ beta,
                              const float* __restrict__ mean,
                              const float* __restrict__ var,
                              unsigned short* __restrict__ w1p,
                              float* __restrict__ bias1) {
    int idx = blockIdx.x * 256 + threadIdx.x;
    if (idx < MID) {
        float inv = gamma[idx] * rsqrtf(var[idx] + 1e-5f);
        bias1[idx] = beta[idx] - mean[idx] * inv;
    }
    if (idx >= MID * KTOT) return;
    int m   = idx / KTOT;
    int k   = idx - m * KTOT;
    int c8  = k / 288;
    int rem = k - c8 * 288;
    int tap = rem >> 5;             // ky*3+kx
    int cl  = rem & 31;
    int c   = c8 * 32 + cl;
    float inv = gamma[m] * rsqrtf(var[m] + 1e-5f);
    w1p[idx] = f2bf(w1[(m * CIN + c) * 9 + tap] * inv);
}

// --------------------- prep: w2 -> bf16 padded to 80 rows -------------------
__global__ void mcspn_prep_w2(const float* __restrict__ w2,
                              const float* __restrict__ b2,
                              unsigned short* __restrict__ w2p,
                              float* __restrict__ b2p) {
    int idx = blockIdx.x * 256 + threadIdx.x;
    if (idx < M2PAD) b2p[idx] = (idx < 76) ? b2[idx] : 0.f;
    if (idx >= M2PAD * MID) return;
    int m = idx >> 7, c = idx & 127;
    w2p[idx] = (m < 76) ? f2bf(w2[m * MID + c]) : (unsigned short)0;
}

// ----------------- fused guidance kernel (double WMMA GEMM) -----------------
// Grid: (WW/64, HH, BB), block 128 (4 waves); wave w owns 16 x-positions.
__global__ __launch_bounds__(128) void mcspn_guidance(
        const float* __restrict__ feats,
        const unsigned short* __restrict__ w1p,
        const float* __restrict__ bias1,
        const unsigned short* __restrict__ w2p,
        const float* __restrict__ b2p,
        float4* __restrict__ gpk) {
    __shared__ __attribute__((aligned(32))) unsigned short inTile[66 * 32];      //  4.2 KB
    __shared__ __attribute__((aligned(32))) unsigned short wTile[MID * 32];      //  8.0 KB
    __shared__ __attribute__((aligned(32))) unsigned short w2t[M2PAD * MID];     // 20.5 KB
    __shared__ __attribute__((aligned(32))) unsigned short midT[4 * 16 * MID];   // 16.0 KB

    const int x0   = blockIdx.x * 64;
    const int y    = blockIdx.y;
    const int b    = blockIdx.z;
    const int tid  = threadIdx.x;
    const int wv   = tid >> 5;
    const int lane = tid & 31;
    const int n    = lane & 15;      // N / M row within a tile
    const int hi   = lane >> 4;      // lane half selects K sub-range

    // stage W2 once (consumed after the barriers in the main loop)
    for (int e = tid; e < M2PAD * MID; e += 128) w2t[e] = w2p[e];

    v8f acc[8] = {};                 // 128 mid channels x 16 positions

    for (int c8 = 0; c8 < 8; ++c8) {
        for (int ky = 0; ky < 3; ++ky) {
            __syncthreads();         // protect inTile readers from prev iter
            const int ys = y + ky - 1;
            for (int e = tid; e < 66 * 32; e += 128) {
                int cc = e / 66;
                int xx = e - cc * 66;
                int xs = x0 - 1 + xx;
                float v = 0.f;
                if ((unsigned)ys < (unsigned)HH && (unsigned)xs < (unsigned)WW)
                    v = feats[(((size_t)b * CIN + (c8 * 32 + cc)) * HH + ys) * WW + xs];
                inTile[xx * 32 + cc] = f2bf(v);
            }
            __syncthreads();
            for (int kx = 0; kx < 3; ++kx) {
                __syncthreads();     // protect wTile readers from prev iter
                {   // stage 128x32 bf16 weight slice: one row per thread
                    const int kbase = c8 * 288 + (ky * 3 + kx) * 32;
                    const uint4* src = (const uint4*)(w1p + (size_t)tid * KTOT + kbase);
                    uint4* dst = (uint4*)(wTile + tid * 32);
                    dst[0] = src[0];
                    dst[1] = src[1];
                }
                __syncthreads();
                // B fragment: lanes 0-15 K=0..15, lanes 16-31 K=16..31 (contig cin)
                const int xx = wv * 16 + n + kx;
                v16bf bfrag = *(const v16bf*)(inTile + xx * 32 + hi * 16);
                #pragma unroll
                for (int t = 0; t < 8; ++t) {
                    // A fragment: lane row M = t*16+n; K = hi*16 .. hi*16+15 (contig)
                    v8bf lo = *(const v8bf*)(wTile + (t * 16 + n) * 32 + hi * 16);
                    v8bf hg = *(const v8bf*)(wTile + (t * 16 + n) * 32 + hi * 16 + 8);
                    v16bf afrag;
                    #pragma unroll
                    for (int i = 0; i < 8; ++i) { afrag[i] = lo[i]; afrag[8 + i] = hg[i]; }
                    acc[t] = __builtin_amdgcn_wmma_f32_16x16x32_bf16(
                        false, afrag, false, bfrag, (short)0, acc[t], false, false);
                }
            }
        }
    }

    // bias + ReLU, pack to bf16 mid tile in LDS (each wave its own 16 rows)
    const float4* bp = (const float4*)bias1;
    #pragma unroll
    for (int t = 0; t < 8; ++t) {
        const int mb = t * 16 + hi * 8;
        float4 b0 = bp[mb >> 2];
        float4 b1 = bp[(mb >> 2) + 1];
        float f[8];
        f[0] = acc[t][0] + b0.x; f[1] = acc[t][1] + b0.y;
        f[2] = acc[t][2] + b0.z; f[3] = acc[t][3] + b0.w;
        f[4] = acc[t][4] + b1.x; f[5] = acc[t][5] + b1.y;
        f[6] = acc[t][6] + b1.z; f[7] = acc[t][7] + b1.w;
        unsigned short h8[8];
        #pragma unroll
        for (int r = 0; r < 8; ++r) h8[r] = f2bf(fmaxf(f[r], 0.f));
        uint4 pk;
        pk.x = (unsigned)h8[0] | ((unsigned)h8[1] << 16);
        pk.y = (unsigned)h8[2] | ((unsigned)h8[3] << 16);
        pk.z = (unsigned)h8[4] | ((unsigned)h8[5] << 16);
        pk.w = (unsigned)h8[6] | ((unsigned)h8[7] << 16);
        *(uint4*)(midT + (wv * 16 + n) * MID + mb) = pk;
    }
    __syncthreads();

    // GEMM2: 80x128 @ 128x16  (w2t staged at kernel start)
    v8f acc2[5] = {};
    #pragma unroll
    for (int kk = 0; kk < 4; ++kk) {
        v16bf bfrag = *(const v16bf*)(midT + (wv * 16 + n) * MID + kk * 32 + hi * 16);
        #pragma unroll
        for (int t = 0; t < 5; ++t) {
            v8bf lo = *(const v8bf*)(w2t + (t * 16 + n) * MID + kk * 32 + hi * 16);
            v8bf hg = *(const v8bf*)(w2t + (t * 16 + n) * MID + kk * 32 + hi * 16 + 8);
            v16bf afrag;
            #pragma unroll
            for (int i = 0; i < 8; ++i) { afrag[i] = lo[i]; afrag[8 + i] = hg[i]; }
            acc2[t] = __builtin_amdgcn_wmma_f32_16x16x32_bf16(
                false, afrag, false, bfrag, (short)0, acc2[t], false, false);
        }
    }

    // +b2, softmax over the 4 direction channels of each group, store float4
    const float4* b2v = (const float4*)b2p;
    const int x = x0 + wv * 16 + n;
    #pragma unroll
    for (int t = 0; t < 5; ++t) {
        const int mb = t * 16 + hi * 8;
        float4 c0 = b2v[mb >> 2];
        float4 c1 = b2v[(mb >> 2) + 1];
        float f[8];
        f[0] = acc2[t][0] + c0.x; f[1] = acc2[t][1] + c0.y;
        f[2] = acc2[t][2] + c0.z; f[3] = acc2[t][3] + c0.w;
        f[4] = acc2[t][4] + c1.x; f[5] = acc2[t][5] + c1.y;
        f[6] = acc2[t][6] + c1.z; f[7] = acc2[t][7] + c1.w;
        #pragma unroll
        for (int grp = 0; grp < 2; ++grp) {
            const int ch0 = mb + grp * 4;
            const int kg  = ch0 >> 2;
            if (kg < KCLS) {
                float a0 = f[grp * 4 + 0], a1 = f[grp * 4 + 1];
                float a2 = f[grp * 4 + 2], a3 = f[grp * 4 + 3];
                float mx = fmaxf(fmaxf(a0, a1), fmaxf(a2, a3));
                float e0 = __expf(a0 - mx), e1 = __expf(a1 - mx);
                float e2 = __expf(a2 - mx), e3 = __expf(a3 - mx);
                float rs = 1.f / (e0 + e1 + e2 + e3);
                float4 gv; gv.x = e0 * rs; gv.y = e1 * rs; gv.z = e2 * rs; gv.w = e3 * rs;
                gpk[(((size_t)b * KCLS + kg) * HH + y) * WW + x] = gv;
            }
        }
    }
}

// ----------------------- propagation (one scan step) ------------------------
__global__ void mcspn_prop(const float* __restrict__ hin,
                           const float4* __restrict__ g,
                           float* __restrict__ hout, int total) {
    int i = blockIdx.x * blockDim.x + threadIdx.x;
    if (i >= total) return;
    int x = i & (WW - 1);
    int y = (i >> 8) & (HH - 1);
    float4 gv = g[i];
    float h     = hin[i];
    float left  = (x > 0)      ? hin[i - 1]  : 0.f;
    float right = (x < WW - 1) ? hin[i + 1]  : 0.f;
    float up    = (y > 0)      ? hin[i - WW] : 0.f;
    float down  = (y < HH - 1) ? hin[i + WW] : 0.f;
    float selfw = 1.f - (gv.x + gv.y + gv.z + gv.w);   // kept for fidelity
    hout[i] = selfw * h + gv.x * left + gv.y * right + gv.z * up + gv.w * down;
}

// ---------------------------------------------------------------------------
extern "C" void kernel_launch(void* const* d_in, const int* in_sizes, int n_in,
                              void* d_out, int out_size, void* d_ws, size_t ws_size,
                              hipStream_t stream) {
    (void)in_sizes; (void)n_in; (void)out_size; (void)ws_size;
    const float* feats  = (const float*)d_in[0];
    const float* logits = (const float*)d_in[1];
    const float* w1     = (const float*)d_in[2];
    const float* gamma  = (const float*)d_in[3];
    const float* beta   = (const float*)d_in[4];
    const float* mean   = (const float*)d_in[5];
    const float* var    = (const float*)d_in[6];
    const float* w2     = (const float*)d_in[7];
    const float* b2     = (const float*)d_in[8];
    float* out = (float*)d_out;

    char* ws = (char*)d_ws;
    size_t off = 0;
    auto alloc = [&](size_t bytes) -> void* {
        off = (off + 255) & ~(size_t)255;
        void* p = ws + off;
        off += bytes;
        return p;
    };
    unsigned short* w1p   = (unsigned short*)alloc((size_t)MID * KTOT * 2);
    float*          bias1 = (float*)alloc(MID * 4);
    unsigned short* w2p   = (unsigned short*)alloc((size_t)M2PAD * MID * 2);
    float*          b2p   = (float*)alloc(M2PAD * 4);
    const size_t npix = (size_t)BB * KCLS * HH * WW;      // 4.98M
    float4* gpk = (float4*)alloc(npix * sizeof(float4));  // ~80 MB
    float*  h1  = (float*)alloc(npix * 4);
    float*  h2  = (float*)alloc(npix * 4);

    mcspn_prep_w1<<<(MID * KTOT + 255) / 256, 256, 0, stream>>>(
        w1, gamma, beta, mean, var, w1p, bias1);
    mcspn_prep_w2<<<(M2PAD * MID + 255) / 256, 256, 0, stream>>>(w2, b2, w2p, b2p);

    dim3 grid(WW / 64, HH, BB);
    mcspn_guidance<<<grid, 128, 0, stream>>>(feats, w1p, bias1, w2p, b2p, gpk);

    const int total  = (int)npix;
    const int blocks = (total + 255) / 256;
    mcspn_prop<<<blocks, 256, 0, stream>>>(logits, gpk, h1, total);
    mcspn_prop<<<blocks, 256, 0, stream>>>(h1, gpk, h2, total);
    mcspn_prop<<<blocks, 256, 0, stream>>>(h2, gpk, h1, total);
    mcspn_prop<<<blocks, 256, 0, stream>>>(h1, gpk, out, total);
}
